// NERBiLSTMCRF_68856915690110
// MI455X (gfx1250) — compile-verified, matrix operand order checked by reading, not run on hardware
//
#include <hip/hip_runtime.h>
#include <hip/hip_bf16.h>

typedef __bf16 bf16_t;
typedef bf16_t v16bf __attribute__((ext_vector_type(16)));
typedef bf16_t v8bf  __attribute__((ext_vector_type(8)));
typedef bf16_t v4bf  __attribute__((ext_vector_type(4)));
typedef float  v8f   __attribute__((ext_vector_type(8)));

#define B_  256
#define S_  512
#define E_  128
#define HH_ 128
#define T_  9

// branch-free activations (no EXEC divergence; robust at +-inf)
__device__ __forceinline__ float sigm(float x) {
  return 1.0f / (1.0f + __expf(-x));
}
__device__ __forceinline__ float tanh_bf(float x) {
  return 1.0f - 2.0f / (__expf(2.0f * x) + 1.0f);
}

// ---------------------------------------------------------------------------
// Kernel 1: pre-swizzle Wih/Whh (f32 [512,128]) into bf16 WMMA B-fragment
// layout: [nt(32)][kstep(4)][lane(32)][elem(16)], and fold bih+bhh.
//   n = nt*16 + (l&15),  k = t*32 + ((l&16)?8:0) + (i<8 ? i : i+8)
// ---------------------------------------------------------------------------
__global__ void prep_kernel(const float* WihF, const float* WhhF,
                            const float* WihB, const float* WhhB,
                            const float* bihF, const float* bhhF,
                            const float* bihB, const float* bhhB,
                            bf16_t* fWihF, bf16_t* fWhhF,
                            bf16_t* fWihB, bf16_t* fWhhB,
                            float* biasF, float* biasB) {
  int idx = blockIdx.x * blockDim.x + threadIdx.x;
  const int NFRAG = 4 * 65536;
  if (idx < NFRAG) {
    int arr = idx >> 16;
    int rem = idx & 65535;
    int i    = rem & 15;
    int lane = (rem >> 4) & 31;
    int t    = (rem >> 9) & 3;
    int nt   = rem >> 11;
    int n = nt * 16 + (lane & 15);
    int k = t * 32 + ((lane & 16) ? 8 : 0) + (i < 8 ? i : i + 8);
    const float* src = (arr == 0) ? WihF : (arr == 1) ? WhhF
                     : (arr == 2) ? WihB : WhhB;
    bf16_t* dst = (arr == 0) ? fWihF : (arr == 1) ? fWhhF
                : (arr == 2) ? fWihB : fWhhB;
    dst[rem] = (bf16_t)src[n * 128 + k];
  } else if (idx < NFRAG + 1024) {
    int j = idx - NFRAG;
    if (j < 512) biasF[j] = bihF[j] + bhhF[j];
    else { int q = j - 512; biasB[q] = bihB[q] + bhhB[q]; }
  }
}

// ---------------------------------------------------------------------------
// Kernel 2: fused bidirectional LSTM recurrence (software-pipelined).
//   blocks 0..15 forward, 16..31 backward; 16 batch rows per block.
//   512 threads = 16 waves; wave w owns gate columns [w*32, w*32+32).
// Per step: g = bias + emb@Wih^T + h@Whh^T via v_wmma_f32_16x16x32_bf16;
// next step's embedding float4 is prefetched during the WMMA phase.
// ---------------------------------------------------------------------------
__global__ __launch_bounds__(512) void lstm_kernel(
    const int* __restrict__ x, const float* __restrict__ emb_table,
    const bf16_t* __restrict__ fWihF, const bf16_t* __restrict__ fWhhF,
    const float* __restrict__ biasF,
    const bf16_t* __restrict__ fWihB, const bf16_t* __restrict__ fWhhB,
    const float* __restrict__ biasB,
    bf16_t* __restrict__ hF, bf16_t* __restrict__ hB) {
  __shared__ bf16_t Aemb[16][128];   // embedding rows (bf16)
  __shared__ bf16_t Ah[16][128];     // recurrent h (bf16)
  __shared__ float  Gm[16][512];     // gate pre-activations
  __shared__ float  Cc[16][128];     // cell state
  __shared__ int    Toks[16][512];   // token ids for this block's rows

  const int tid = threadIdx.x;
  const int dir = blockIdx.x >> 4;
  const int b0  = (blockIdx.x & 15) * 16;
  const bf16_t* fWih = dir ? fWihB : fWihF;
  const bf16_t* fWhh = dir ? fWhhB : fWhhF;
  const float*  bias = dir ? biasB : biasF;
  bf16_t* hOut = dir ? hB : hF;

  const int w    = tid >> 5;
  const int lane = tid & 31;
  const int rsel = lane >> 4;     // 0 or 1 (lane half)
  const int mrow = lane & 15;
  const int row  = tid >> 5;      // 0..15 (elementwise / gather mapping)
  const int j0   = (tid & 31) * 4;

  // preload all token ids (coalesced), zero-init h0/c0
  for (int idx = tid; idx < 16 * 512; idx += 512) {
    int r = idx >> 9, ss = idx & 511;
    Toks[r][ss] = x[(b0 + r) * S_ + ss];
  }
  for (int p = tid; p < 16 * 128; p += 512) {
    ((bf16_t*)Ah)[p] = (bf16_t)0.0f;
    ((float*)Cc)[p]  = 0.0f;
  }

  // loop-invariant weight fragments -> registers for all 512 steps
  v16bf wih[2][4], whh[2][4];
  float biasv[2];
#pragma unroll
  for (int q = 0; q < 2; ++q) {
    int nt = w * 2 + q;
    biasv[q] = bias[nt * 16 + mrow];
#pragma unroll
    for (int t = 0; t < 4; ++t) {
      size_t off = ((size_t)(nt * 4 + t) * 32 + lane) * 16;
      wih[q][t] = *(const v16bf*)(fWih + off);
      whh[q][t] = *(const v16bf*)(fWhh + off);
    }
  }
  __syncthreads();

  // prologue: fetch step-0 embedding and stage into LDS
  {
    int s0 = dir ? (S_ - 1) : 0;
    float4 ev = *(const float4*)(emb_table + (size_t)Toks[row][s0] * E_ + j0);
    Aemb[row][j0 + 0] = (bf16_t)ev.x;
    Aemb[row][j0 + 1] = (bf16_t)ev.y;
    Aemb[row][j0 + 2] = (bf16_t)ev.z;
    Aemb[row][j0 + 3] = (bf16_t)ev.w;
  }
  __syncthreads();

  for (int it = 0; it < S_; ++it) {
    const int s = dir ? (S_ - 1 - it) : it;

    // build A fragments (16x32 bf16 tiles) from LDS
    v16bf ae[4], ah[4];
#pragma unroll
    for (int t = 0; t < 4; ++t) {
      int k0 = t * 32 + rsel * 8;
      v8bf e0 = *(const v8bf*)(&Aemb[mrow][k0]);
      v8bf e1 = *(const v8bf*)(&Aemb[mrow][k0 + 16]);
      v8bf h0 = *(const v8bf*)(&Ah[mrow][k0]);
      v8bf h1 = *(const v8bf*)(&Ah[mrow][k0 + 16]);
#pragma unroll
      for (int i = 0; i < 8; ++i) {
        ae[t][i] = e0[i]; ae[t][i + 8] = e1[i];
        ah[t][i] = h0[i]; ah[t][i + 8] = h1[i];
      }
    }

    // prefetch next step's embedding (waited only at the staging store below)
    int itn = (it + 1 < S_) ? (it + 1) : it;
    int sn  = dir ? (S_ - 1 - itn) : itn;
    float4 evn = *(const float4*)(emb_table + (size_t)Toks[row][sn] * E_ + j0);

    // g = bias + emb@Wih^T + h@Whh^T  (WMMA bf16, f32 accumulate)
#pragma unroll
    for (int q = 0; q < 2; ++q) {
      v8f acc;
#pragma unroll
      for (int v = 0; v < 8; ++v) acc[v] = biasv[q];
#pragma unroll
      for (int t = 0; t < 4; ++t) {
        acc = __builtin_amdgcn_wmma_f32_16x16x32_bf16(
            false, ae[t], false, wih[q][t], (short)0, acc, false, false);
        acc = __builtin_amdgcn_wmma_f32_16x16x32_bf16(
            false, ah[t], false, whh[q][t], (short)0, acc, false, false);
      }
      int n = (w * 2 + q) * 16 + mrow;
      int rbase = rsel * 8;
#pragma unroll
      for (int v = 0; v < 8; ++v) Gm[rbase + v][n] = acc[v];
    }
    __syncthreads();

    // LSTM cell update (torch gate order i,f,g,o), branch-free activations
    {
      v4bf hv;
#pragma unroll
      for (int p = 0; p < 4; ++p) {
        int j = j0 + p;
        float iv = Gm[row][j];
        float fv = Gm[row][128 + j];
        float gv = Gm[row][256 + j];
        float ov = Gm[row][384 + j];
        float c  = Cc[row][j];
        c = sigm(fv) * c + sigm(iv) * tanh_bf(gv);
        float h = sigm(ov) * tanh_bf(c);
        Cc[row][j] = c;
        Ah[row][j] = (bf16_t)h;
        hv[p] = (bf16_t)h;
      }
      *(v4bf*)(hOut + ((size_t)s * B_ + (b0 + row)) * HH_ + j0) = hv;
      // stage next step's embedding (Aemb(s) already consumed pre-barrier)
      Aemb[row][j0 + 0] = (bf16_t)evn.x;
      Aemb[row][j0 + 1] = (bf16_t)evn.y;
      Aemb[row][j0 + 2] = (bf16_t)evn.z;
      Aemb[row][j0 + 3] = (bf16_t)evn.w;
    }
    __syncthreads();
  }
}

// ---------------------------------------------------------------------------
// Kernel 3: emissions em[b,s,t] = ([hf|hb] @ W_out^T + b_out) * mask
// ---------------------------------------------------------------------------
__global__ void emission_kernel(const bf16_t* __restrict__ hF,
                                const bf16_t* __restrict__ hB,
                                const float* __restrict__ W_out,
                                const float* __restrict__ b_out,
                                const float* __restrict__ mask,
                                float* __restrict__ em) {
  int pos = blockIdx.x * blockDim.x + threadIdx.x;
  if (pos >= B_ * S_) return;
  int b = pos >> 9;
  int s = pos & 511;
  const bf16_t* pf = hF + ((size_t)s * B_ + b) * HH_;
  const bf16_t* pb = hB + ((size_t)s * B_ + b) * HH_;
  float acc[T_];
#pragma unroll
  for (int t = 0; t < T_; ++t) acc[t] = b_out[t];
  for (int jc = 0; jc < HH_; jc += 8) {
    v8bf hf8 = *(const v8bf*)(pf + jc);
    v8bf hb8 = *(const v8bf*)(pb + jc);
#pragma unroll
    for (int i = 0; i < 8; ++i) {
      float hf = (float)hf8[i];
      float hb = (float)hb8[i];
      int j = jc + i;
#pragma unroll
      for (int t = 0; t < T_; ++t)
        acc[t] += hf * W_out[t * 256 + j] + hb * W_out[t * 256 + 128 + j];
    }
  }
  float m = mask[b * S_ + s];
#pragma unroll
  for (int t = 0; t < T_; ++t) em[(size_t)pos * T_ + t] = acc[t] * m;
}

// ---------------------------------------------------------------------------
// Kernel 4: Viterbi decode, one lane per batch element.
// ---------------------------------------------------------------------------
__global__ void viterbi_kernel(const float* __restrict__ em,
                               const float* __restrict__ mask,
                               const float* __restrict__ transitions,
                               const float* __restrict__ start_trans,
                               const float* __restrict__ end_trans,
                               unsigned char* __restrict__ bp,
                               float* __restrict__ out) {
  int b = threadIdx.x;
  float tr[T_ * T_];
#pragma unroll
  for (int k = 0; k < T_ * T_; ++k) tr[k] = transitions[k];
  float score[T_];
#pragma unroll
  for (int t = 0; t < T_; ++t)
    score[t] = start_trans[t] + em[((size_t)b * S_) * T_ + t];

  for (int s = 1; s < S_; ++s) {
    const float* e = em + ((size_t)b * S_ + s) * T_;
    bool live = mask[b * S_ + s] > 0.0f;
    float ns[T_]; int bpv[T_];
#pragma unroll
    for (int cur = 0; cur < T_; ++cur) {
      float best = score[0] + tr[cur];
      int arg = 0;
#pragma unroll
      for (int prev = 1; prev < T_; ++prev) {
        float v = score[prev] + tr[prev * T_ + cur];
        if (v > best) { best = v; arg = prev; }   // strict > == jnp first-max
      }
      ns[cur]  = best + e[cur];
      bpv[cur] = arg;
    }
    unsigned char* brow = bp + ((size_t)s * B_ + b) * T_;
#pragma unroll
    for (int cur = 0; cur < T_; ++cur) {
      if (live) score[cur] = ns[cur];
      brow[cur] = (unsigned char)(live ? bpv[cur] : cur);
    }
  }

  int best_t = 0;
  float best_v = score[0] + end_trans[0];
#pragma unroll
  for (int t = 1; t < T_; ++t) {
    float v = score[t] + end_trans[t];
    if (v > best_v) { best_v = v; best_t = t; }
  }
  float* tags   = out;             // [B,S] as float
  float* scores = out + B_ * S_;   // [B]
  scores[b] = best_v;
  int cur = best_t;
  tags[b * S_ + (S_ - 1)] = (float)cur;
  for (int s = S_ - 1; s >= 1; --s) {
    cur = bp[((size_t)s * B_ + b) * T_ + cur];
    tags[b * S_ + (s - 1)] = (float)cur;
  }
}

// ---------------------------------------------------------------------------
extern "C" void kernel_launch(void* const* d_in, const int* in_sizes, int n_in,
                              void* d_out, int out_size, void* d_ws, size_t ws_size,
                              hipStream_t stream) {
  (void)in_sizes; (void)n_in; (void)out_size; (void)ws_size;
  const int*   x         = (const int*)d_in[0];
  const float* mask      = (const float*)d_in[1];
  const float* emb_table = (const float*)d_in[2];
  const float* Wih_f = (const float*)d_in[3];
  const float* Whh_f = (const float*)d_in[4];
  const float* bih_f = (const float*)d_in[5];
  const float* bhh_f = (const float*)d_in[6];
  const float* Wih_b = (const float*)d_in[7];
  const float* Whh_b = (const float*)d_in[8];
  const float* bih_b = (const float*)d_in[9];
  const float* bhh_b = (const float*)d_in[10];
  const float* W_out = (const float*)d_in[11];
  const float* b_out = (const float*)d_in[12];
  const float* transitions = (const float*)d_in[13];
  const float* start_trans = (const float*)d_in[14];
  const float* end_trans   = (const float*)d_in[15];

  char* ws = (char*)d_ws;
  bf16_t* fWihF = (bf16_t*)(ws + 0);
  bf16_t* fWhhF = (bf16_t*)(ws + 131072);
  bf16_t* fWihB = (bf16_t*)(ws + 262144);
  bf16_t* fWhhB = (bf16_t*)(ws + 393216);
  float*  biasF = (float*)(ws + 524288);
  float*  biasB = (float*)(ws + 526336);
  bf16_t* hF    = (bf16_t*)(ws + 528384);
  bf16_t* hB    = (bf16_t*)(ws + 528384 + 33554432);
  float*  em    = (float*)(ws + 528384 + 2u * 33554432);
  unsigned char* bp =
      (unsigned char*)(ws + 528384 + 2u * 33554432 + 4718592);

  prep_kernel<<<1028, 256, 0, stream>>>(Wih_f, Whh_f, Wih_b, Whh_b,
      bih_f, bhh_f, bih_b, bhh_b, fWihF, fWhhF, fWihB, fWhhB, biasF, biasB);
  lstm_kernel<<<32, 512, 0, stream>>>(x, emb_table, fWihF, fWhhF, biasF,
      fWihB, fWhhB, biasB, hF, hB);
  emission_kernel<<<(B_ * S_) / 256, 256, 0, stream>>>(hF, hB, W_out, b_out,
                                                       mask, em);
  viterbi_kernel<<<1, 256, 0, stream>>>(em, mask, transitions, start_trans,
                                        end_trans, bp, (float*)d_out);
}
